// AFM_5944234738104
// MI455X (gfx1250) — compile-verified, matrix operand order checked by reading, not run on hardware
//
#include <hip/hip_runtime.h>
#include <hip/hip_bf16.h>

typedef __attribute__((ext_vector_type(16))) __bf16 v16bf;
typedef __attribute__((ext_vector_type(8)))  float  v8f;

#define NFIELD 39
#define NCONT  13
#define NCATE  26
#define EMB    40
#define NPAIR  741          // 39*38/2
#define NTILE  47           // ceil(741/16)
#define ADIM   8
#define NTHREADS 256        // 8 wave32
#define NWAVE  8
#define PGRP   6            // pooled reduction groups (6*40=240 lanes active)

__global__ __launch_bounds__(NTHREADS)
void afm_fused_kernel(const float* __restrict__ conts,
                      const int*   __restrict__ cates,
                      const float* __restrict__ emb,
                      const float* __restrict__ attn_W,
                      const float* __restrict__ attn_b,
                      const float* __restrict__ proj_W,
                      const float* __restrict__ fc_W,
                      const float* __restrict__ fc_b,
                      float* __restrict__ out)
{
    __shared__ float          sX[NFIELD * EMB];     // 39x40 field embeddings
    __shared__ float          sLogit[NTILE * 16];   // logits, then exp()
    __shared__ unsigned short sRow[NPAIR];
    __shared__ unsigned short sCol[NPAIR];
    __shared__ float          sRed[NTHREADS];
    __shared__ float          sPool[PGRP * EMB];

    const int t    = threadIdx.x;
    const int b    = blockIdx.x;
    const int lane = t & 31;
    const int wave = t >> 5;

    // ---- Phase 0a: gather x = [cont_emb ; cate_emb] into LDS -------------
    for (int i = t; i < NFIELD * EMB; i += NTHREADS) {
        int r = i / EMB, d = i - r * EMB;
        float v;
        if (r < NCONT) v = emb[r * EMB + d] * conts[b * NCONT + r];
        else           v = emb[(size_t)cates[b * NCATE + (r - NCONT)] * EMB + d];
        sX[i] = v;
    }
    // ---- Phase 0b: triu(k=1) pair index tables ---------------------------
    for (int p = t; p < NPAIR; p += NTHREADS) {
        int r = 0, rem = p, cnt = NFIELD - 1;
        while (rem >= cnt) { rem -= cnt; ++r; --cnt; }
        sRow[p] = (unsigned short)r;
        sCol[p] = (unsigned short)(r + 1 + rem);
    }
    __syncthreads();

    // ---- Loop-invariant A fragment: attn_W as M=16(8 real) x K=64(40 real)
    // 16-bit A layout (ISA 7.12.2): lane L holds row m=L%16,
    // K in {kb..kb+7, kb+16..kb+23}, kb=(L<16)?0:8.
    const int m_a = lane & 15;          // attention row this lane supplies
    const int kb  = (lane < 16) ? 0 : 8;
    v16bf Aw0 = {}, Aw1 = {};
#pragma unroll
    for (int j = 0; j < 8; ++j) {
        int k0 = kb + j, k1 = kb + 16 + j;                   // k0<=23, k1<=31
        float a00 = (m_a < ADIM) ? attn_W[m_a * EMB + k0] : 0.f;
        float a01 = (m_a < ADIM) ? attn_W[m_a * EMB + k1] : 0.f;
        float a10 = (m_a < ADIM && (32 + k0) < EMB) ? attn_W[m_a * EMB + 32 + k0] : 0.f;
        Aw0[j]     = (__bf16)a00;
        Aw0[8 + j] = (__bf16)a01;
        Aw1[j]     = (__bf16)a10;      // K=32..39 real, 40..47 zero-pad
        Aw1[8 + j] = (__bf16)0.f;      // K=48..63 zero-pad
    }

    // Uniform epilogue constants (scalarized by compiler): all 8 attn dims
    float bias[ADIM], proj[ADIM];
#pragma unroll
    for (int r = 0; r < ADIM; ++r) { bias[r] = attn_b[r]; proj[r] = proj_W[r]; }

    // ---- Phase 1: logits via C = attn_W @ ewp^T  (V_WMMA_F32_16X16X32_BF16)
    // B fragment = ewp^T, K=64(40 real) x N=16 pairs: lane L holds pair
    // n=L%16, same K-chunk split as A. After WMMA, lane n<16 owns ALL 8
    // attention features of pair n in c[0..7] -> in-lane logit, no shuffles.
    for (int tile = wave; tile < NTILE; tile += NWAVE) {
        const int p = tile * 16 + (lane & 15);
        v16bf E0 = {}, E1 = {};
        if (p < NPAIR) {
            const float* xr = &sX[(int)sRow[p] * EMB];
            const float* xc = &sX[(int)sCol[p] * EMB];
#pragma unroll
            for (int j = 0; j < 8; ++j) {
                int k0 = kb + j, k1 = kb + 16 + j;
                E0[j]     = (__bf16)(xr[k0] * xc[k0]);
                E0[8 + j] = (__bf16)(xr[k1] * xc[k1]);
                int k2 = 32 + k0;
                E1[j]     = (k2 < EMB) ? (__bf16)(xr[k2] * xc[k2]) : (__bf16)0.f;
                E1[8 + j] = (__bf16)0.f;
            }
        }
        v8f c = {};
        c = __builtin_amdgcn_wmma_f32_16x16x32_bf16(false, Aw0, false, E0,
                                                    (short)0, c, false, false);
        c = __builtin_amdgcn_wmma_f32_16x16x32_bf16(false, Aw1, false, E1,
                                                    (short)0, c, false, false);

        if (lane < 16 && p < NPAIR) {   // lanes 16-31 hold pad rows 8..15
            float lg = 0.f;
#pragma unroll
            for (int r = 0; r < ADIM; ++r)
                lg += fmaxf(c[r] + bias[r], 0.f) * proj[r];
            sLogit[p] = lg;
        }
    }
    __syncthreads();

    // ---- Phase 2: softmax over 741 pairs (block reduction) ---------------
    float lm = -__builtin_inff();
    for (int p = t; p < NPAIR; p += NTHREADS) lm = fmaxf(lm, sLogit[p]);
    sRed[t] = lm;
    __syncthreads();
    for (int s = NTHREADS / 2; s > 0; s >>= 1) {
        if (t < s) sRed[t] = fmaxf(sRed[t], sRed[t + s]);
        __syncthreads();
    }
    const float mx = sRed[0];
    __syncthreads();

    float ls = 0.f;
    for (int p = t; p < NPAIR; p += NTHREADS) {
        float e = __expf(sLogit[p] - mx);
        sLogit[p] = e;                            // store unnormalized weights
        ls += e;
    }
    sRed[t] = ls;
    __syncthreads();
    for (int s = NTHREADS / 2; s > 0; s >>= 1) {
        if (t < s) sRed[t] += sRed[t + s];
        __syncthreads();
    }
    const float inv = 1.f / sRed[0];
    __syncthreads();

    // ---- Phase 3: pooled[d] = inv * sum_p e_p * ewp[p][d] (recompute ewp)
    if (t < PGRP * EMB) {
        const int d = t % EMB;
        const int g = t / EMB;
        float pp = 0.f;
        for (int p = g; p < NPAIR; p += PGRP) {
            float e = sLogit[p];
            pp += e * sX[(int)sRow[p] * EMB + d] * sX[(int)sCol[p] * EMB + d];
        }
        sPool[t] = pp;
    }
    __syncthreads();

    // ---- Phase 4: out = sigmoid(pooled . fc_W + fc_b) --------------------
    if (t < EMB) {
        float s = 0.f;
#pragma unroll
        for (int g = 0; g < PGRP; ++g) s += sPool[g * EMB + t];
        sRed[t] = s * inv * fc_W[t];
    }
    __syncthreads();
    if (t == 0) {
        float s = fc_b[0];
        for (int d = 0; d < EMB; ++d) s += sRed[d];
        out[b] = 1.f / (1.f + __expf(-s));
    }
}

extern "C" void kernel_launch(void* const* d_in, const int* in_sizes, int n_in,
                              void* d_out, int out_size, void* d_ws, size_t ws_size,
                              hipStream_t stream) {
    // setup_inputs() order: conts, cates, combs, emb_table, attn_W, attn_b,
    //                       proj_W, fc_W, fc_b   (combs unused by the math)
    const float* conts  = (const float*)d_in[0];
    const int*   cates  = (const int*)  d_in[1];
    const float* emb    = (const float*)d_in[3];
    const float* attn_W = (const float*)d_in[4];
    const float* attn_b = (const float*)d_in[5];
    const float* proj_W = (const float*)d_in[6];
    const float* fc_W   = (const float*)d_in[7];
    const float* fc_b   = (const float*)d_in[8];
    float* out = (float*)d_out;

    const int batch = in_sizes[0] / NCONT;   // 4096
    afm_fused_kernel<<<batch, NTHREADS, 0, stream>>>(
        conts, cates, emb, attn_W, attn_b, proj_W, fc_W, fc_b, out);
}
